// ChunkKVCompressor_1211180777499
// MI455X (gfx1250) — matrix operand order repaired and checked by reading, not compiled, not run on hardware
//
#include <hip/hip_runtime.h>
#include <hip/hip_bf16.h>

// ---------------------------------------------------------------------------
// Problem constants (fixed by reference setup_inputs)
// ---------------------------------------------------------------------------
#define B_SZ       8
#define T_SZ       4096
#define D_SZ       1024          // HIDDEN
#define H_SZ       512           // HIDDEN/2
#define L_SZ       64            // CHUNK_LEN
#define NC_SZ      64            // T / L
#define KEEP_SZ    32            // chunks kept per batch
#define LDS_STRIDE 1040          // bf16 elems per LDS row (1024 + 16 pad)
#define CHUNK_ELEMS (L_SZ * D_SZ)                 // 65536 floats per chunk
#define CV_OFS     ((size_t)B_SZ * KEEP_SZ * CHUNK_ELEMS)  // 16,777,216

typedef __attribute__((ext_vector_type(16))) __bf16 v16bf;
typedef __attribute__((ext_vector_type(8)))  float  v8f;

__device__ __forceinline__ unsigned short f2bf(float f) {
    unsigned int u = __float_as_uint(f);
    u += 0x7fffu + ((u >> 16) & 1u);     // round-to-nearest-even
    return (unsigned short)(u >> 16);
}
__device__ __forceinline__ unsigned int pack2(float a, float b) {
    return (unsigned int)f2bf(a) | ((unsigned int)f2bf(b) << 16);
}

// ---------------------------------------------------------------------------
// Stage 0: one-shot W1 fp32 -> bf16 repack into WMMA B-fragment order.
//   Fragment (ct, ks): 32 lanes x 8 dwords; lane -> N = ct*16 + (lane&15),
//   element s -> K = ks*32 + 8*(lane>>4) + s (s<8), ks*32 + 16 + 8*(lane>>4) + s-8 (s>=8).  grid = 128 x 256 threads.
// ---------------------------------------------------------------------------
__global__ void __launch_bounds__(256)
w1conv_kernel(const float* __restrict__ W1, uint4* __restrict__ W1b)
{
    const int g    = blockIdx.x * 256 + threadIdx.x;   // 0 .. 32767
    const int ct   = g >> 10;
    const int ks   = (g >> 5) & 31;
    const int lane = g & 31;
    const float* wp = W1 + (ks * 32 + (lane >> 4) * 8) * H_SZ + ct * 16 + (lane & 15);
    float4 f0 = make_float4(wp[0 * H_SZ], wp[1 * H_SZ], wp[2 * H_SZ], wp[3 * H_SZ]);
    float4 f1 = make_float4(wp[4 * H_SZ], wp[5 * H_SZ], wp[6 * H_SZ], wp[7 * H_SZ]);
    float4 f2 = make_float4(wp[16 * H_SZ], wp[17 * H_SZ], wp[18 * H_SZ], wp[19 * H_SZ]);
    float4 f3 = make_float4(wp[20 * H_SZ], wp[21 * H_SZ], wp[22 * H_SZ], wp[23 * H_SZ]);
    uint4 q0, q1;
    q0.x = pack2(f0.x, f0.y); q0.y = pack2(f0.z, f0.w);
    q0.z = pack2(f1.x, f1.y); q0.w = pack2(f1.z, f1.w);
    q1.x = pack2(f2.x, f2.y); q1.y = pack2(f2.z, f2.w);
    q1.z = pack2(f3.x, f3.y); q1.w = pack2(f3.z, f3.w);
    W1b[(size_t)g * 2 + 0] = q0;
    W1b[(size_t)g * 2 + 1] = q1;
}

// ---------------------------------------------------------------------------
// Stage 1: per-chunk score via bf16 WMMA.
//   grid.x = B*NC (512), block = 256 (8 waves), dynamic LDS.
//   2-wide column-tile register blocking: each A fragment feeds 2 wmmas
//   (halves LDS read amplification), double-buffered A and B.
// ---------------------------------------------------------------------------
__global__ void __launch_bounds__(256)
score_kernel(const float* __restrict__ keys, const float* __restrict__ vals,
             const uint4* __restrict__ W1b,  const float* __restrict__ b1,
             const float* __restrict__ W2,   const float* __restrict__ b2,
             float* __restrict__ chunk_scores)
{
    extern __shared__ unsigned char smem[];
    unsigned short* sA     = (unsigned short*)smem;                   // [64][LDS_STRIDE] bf16
    float*          sScore = (float*)(smem + L_SZ * LDS_STRIDE * 2);  // [64] token scores

    const int chunk = blockIdx.x;                 // flat b*NC + n
    const int tid   = threadIdx.x;
    const size_t gbase = (size_t)chunk * CHUNK_ELEMS;   // chunk contiguous in memory

    if (tid < L_SZ) sScore[tid] = 0.0f;

    // ---- stage combined = (K+V)*0.5 as bf16 in LDS (b128 path) ----
    for (int e = tid * 4; e < CHUNK_ELEMS; e += 256 * 4) {
        float4 k4 = *(const float4*)(keys + gbase + e);
        float4 v4 = *(const float4*)(vals + gbase + e);
        float4 c4;
        c4.x = (k4.x + v4.x) * 0.5f;  c4.y = (k4.y + v4.y) * 0.5f;
        c4.z = (k4.z + v4.z) * 0.5f;  c4.w = (k4.w + v4.w) * 0.5f;
        const int row = e >> 10, col = e & (D_SZ - 1);
        uint2 p; p.x = pack2(c4.x, c4.y); p.y = pack2(c4.z, c4.w);
        *(uint2*)(sA + row * LDS_STRIDE + col) = p;
    }
    __syncthreads();

    const int wave = tid >> 5;
    const int lane = tid & 31;
    const int lm   = lane & 15;     // M index (A/C) or N index (B/C)
    const int lh   = lane >> 4;     // K-half select (A), C row-half select

    union AFrag { v16bf v; float4 f[2]; };
    union BFrag { v16bf v; uint4  q[2]; };

    float p[32];                    // per-row score partials, carried across passes
    #pragma unroll
    for (int i = 0; i < 32; ++i) p[i] = 0.0f;

    // Each wave owns ct = {wave, wave+8, wave+16, wave+24}; process 2 at a time
    // so one A-fragment load feeds 2 wmmas.
    for (int cti2 = 0; cti2 < 2; ++cti2) {
        const int ct0 = wave + (cti2 * 2 + 0) * 8;
        const int ct1 = wave + (cti2 * 2 + 1) * 8;
        const float b1v0 = b1[ct0 * 16 + lm], w2v0 = W2[ct0 * 16 + lm];
        const float b1v1 = b1[ct1 * 16 + lm], w2v1 = W2[ct1 * 16 + lm];
        const uint4* __restrict__ wf0 = W1b + (((size_t)ct0 * 32) * 32 + lane) * 2;
        const uint4* __restrict__ wf1 = W1b + (((size_t)ct1 * 32) * 32 + lane) * 2;

        v8f acc[2][4] = {};                      // [ct pair][row tile]
        AFrag a0[4], a1[4];
        BFrag bb0[2], bb1[2];

        auto loadA = [&](AFrag* a, int kb) {
            #pragma unroll
            for (int rt = 0; rt < 4; ++rt) {
                const unsigned short* ap =
                    sA + (rt * 16 + lm) * LDS_STRIDE + kb + lh * 8;
                a[rt].f[0] = *(const float4*)(ap);        // K = kb + lh*8 + 0..7
                a[rt].f[1] = *(const float4*)(ap + 16);   // K = kb + 16 + lh*8 + 0..7
            }
        };
        auto loadB = [&](BFrag* b, int ksx) {
            const uint4* q0 = wf0 + (size_t)ksx * 64;     // 32 lanes * 2 uint4 per frag
            const uint4* q1 = wf1 + (size_t)ksx * 64;
            b[0].q[0] = q0[0]; b[0].q[1] = q0[1];
            b[1].q[0] = q1[0]; b[1].q[1] = q1[1];
        };
        auto mm8 = [&](AFrag* a, BFrag* b) {
            #pragma unroll
            for (int cc = 0; cc < 2; ++cc)
                #pragma unroll
                for (int rt = 0; rt < 4; ++rt)
                    acc[cc][rt] = __builtin_amdgcn_wmma_f32_16x16x32_bf16(
                        false, a[rt].v, false, b[cc].v, (short)0,
                        acc[cc][rt], false, false);
        };

        // ---- K loop, software-pipelined double buffer (2 k-steps/iter) ----
        loadB(bb0, 0);
        loadA(a0, 0);
        for (int ks2 = 0; ks2 < 16; ++ks2) {
            const int k1 = ks2 * 2 + 1;              // always <= 31
            loadB(bb1, k1);
            loadA(a1, k1 * 32);
            mm8(a0, bb0);
            const int k2 = (ks2 * 2 + 2) & 31;       // wraps harmlessly on last iter
            loadB(bb0, k2);
            loadA(a0, k2 * 32);
            mm8(a1, bb1);
        }

        // ---- fused epilogue into registers: +b1, relu, *W2 ----
        #pragma unroll
        for (int rt = 0; rt < 4; ++rt) {
            #pragma unroll
            for (int j = 0; j < 8; ++j) {
                float h0 = acc[0][rt][j] + b1v0;
                float h1 = acc[1][rt][j] + b1v1;
                p[rt * 8 + j] += fmaxf(h0, 0.0f) * w2v0
                               + fmaxf(h1, 0.0f) * w2v1;
            }
        }
    }

    // ---- one batch of fire-and-forget LDS float atomics (no shuffles) ----
    #pragma unroll
    for (int rt = 0; rt < 4; ++rt) {
        #pragma unroll
        for (int j = 0; j < 8; ++j)
            atomicAdd(&sScore[rt * 16 + lh * 8 + j], p[rt * 8 + j]);
    }
    __syncthreads();

    if (wave == 0) {
        float s = sScore[lane] + sScore[lane + 32];
        #pragma unroll
        for (int m = 16; m >= 1; m >>= 1) s += __shfl_xor(s, m, 32);
        if (lane == 0)
            chunk_scores[chunk] = s * (1.0f / L_SZ) + b2[0];
    }
}

// ---------------------------------------------------------------------------
// Stage 2: top-K with jax tie-break (lower index wins), ordered compaction
//   grid = B, block = NC (64 threads)
// ---------------------------------------------------------------------------
__global__ void topk_kernel(const float* __restrict__ chunk_scores,
                            int* __restrict__ idx_out)
{
    __shared__ float s[NC_SZ];
    __shared__ int   keep[NC_SZ];
    const int b = blockIdx.x, t = threadIdx.x;
    s[t] = chunk_scores[b * NC_SZ + t];
    __syncthreads();
    const float mys = s[t];
    int rank = 0;
    for (int j = 0; j < NC_SZ; ++j) {
        const float sj = s[j];
        rank += (sj > mys) || (sj == mys && j < t);
    }
    keep[t] = (rank < KEEP_SZ) ? 1 : 0;
    __syncthreads();
    if (keep[t]) {
        int pos = 0;
        for (int j = 0; j < t; ++j) pos += keep[j];
        idx_out[b * KEEP_SZ + pos] = t;     // ascending index order by construction
    }
}

// ---------------------------------------------------------------------------
// Stage 3: gather kept chunks -> d_out = [ck | cv], float4 streaming copy
//   grid = (B*KEEP, 8), block = 256
// ---------------------------------------------------------------------------
__global__ void __launch_bounds__(256)
gather_kernel(const float* __restrict__ keys, const float* __restrict__ vals,
              const int* __restrict__ idx, float* __restrict__ out)
{
    const int g  = blockIdx.x;            // b*KEEP + kk
    const int b  = g / KEEP_SZ;
    const int c  = idx[g];
    const size_t src = ((size_t)b * NC_SZ + c) * CHUNK_ELEMS;
    const size_t dst = (size_t)g * CHUNK_ELEMS;
    const int slice = blockIdx.y;         // 8 slices x 8192 floats
    const int e0 = slice * 8192 + threadIdx.x * 4;
    const int e1 = (slice + 1) * 8192;
    for (int e = e0; e < e1; e += 256 * 4) {
        *(float4*)(out + dst + e)          = *(const float4*)(keys + src + e);
        *(float4*)(out + CV_OFS + dst + e) = *(const float4*)(vals + src + e);
    }
}

// ---------------------------------------------------------------------------
extern "C" void kernel_launch(void* const* d_in, const int* in_sizes, int n_in,
                              void* d_out, int out_size, void* d_ws, size_t ws_size,
                              hipStream_t stream)
{
    const float* keys   = (const float*)d_in[0];
    const float* values = (const float*)d_in[1];
    const float* W1     = (const float*)d_in[2];
    const float* b1     = (const float*)d_in[3];
    const float* W2     = (const float*)d_in[4];
    const float* b2     = (const float*)d_in[5];

    float* scores = (float*)d_ws;                                        // [512] f32
    int*   idx    = (int*)((char*)d_ws + 2048);                          // [256] i32
    uint4* W1b    = (uint4*)((char*)d_ws + 4096);                        // 1 MB bf16 fragments

    const size_t shmem = (size_t)L_SZ * LDS_STRIDE * sizeof(unsigned short)
                       + (size_t)L_SZ * sizeof(float);                   // ~130.3 KB

    w1conv_kernel<<<dim3(128), dim3(256), 0, stream>>>(W1, W1b);
    score_kernel<<<dim3(B_SZ * NC_SZ), dim3(256), shmem, stream>>>(
        keys, values, W1b, b1, W2, b2, scores);
    topk_kernel<<<dim3(B_SZ), dim3(NC_SZ), 0, stream>>>(scores, idx);
    gather_kernel<<<dim3(B_SZ * KEEP_SZ, 8), dim3(256), 0, stream>>>(
        keys, values, idx, (float*)d_out);
}
